// AM_E_GCL_58463094833684
// MI455X (gfx1250) — matrix operand order repaired and checked by reading, not compiled, not run on hardware
//
#include <hip/hip_runtime.h>
#include <hip/hip_bf16.h>

typedef __attribute__((ext_vector_type(16))) _Float16 v16h;
typedef __attribute__((ext_vector_type(8)))  _Float16 v8h;
typedef __attribute__((ext_vector_type(8)))  float    v8f;

#define HID 128
#define CCH 14

__device__ inline float siluf(float v) { return v / (1.0f + __expf(-v)); }

// ---- WMMA fragment loaders (contiguous k = 16*(lane>=16) + s order, wave32) ----

// A 16x32 f16 row-major, stride lda (halves). Per-lane fragment = two 16B
// contiguous chunks: halves 0..7 = K=koff+0..7, halves 8..15 = K=koff+8..15,
// koff = (lane>=16)*16. Assembled via union so regalloc avoids v_mov packing.
__device__ inline v16h load_A_rm(const _Float16* __restrict__ buf, int lda, int k0, int lane) {
    int m = lane & 15;
    int koff = (lane >> 4) * 16;
    const _Float16* p = buf + m * lda + k0 + koff;
    union { v16h v; v8h h[2]; } u;
    u.h[0] = *(const v8h*)p;
    u.h[1] = *(const v8h*)(p + 8);
    return u.v;
}

// B 32x16 f16 stored TRANSPOSED in LDS as [N][32] row-major: one aligned 32B
// vector load per lane (halves h = K = (lane>=16)*16 + h, col n = lane&15).
__device__ inline v16h load_Bt32(const _Float16* __restrict__ buf, int lane) {
    int n = lane & 15;
    int kb = (lane >> 4) * 16;
    return *(const v16h*)(buf + n * 32 + kb);
}

// GEMM: D[16][16-tile at n0] += A(16xK LDS, stride lda) * W^T.
// W is f16 [out][K] row-major (B[k][n] = W[n0+n][k]); per-lane B tile is 16
// contiguous halves -> one aligned 32B global vector load.
template <int KTOT>
__device__ inline v8f gemm_wmma(const _Float16* __restrict__ As, int lda,
                                const _Float16* __restrict__ W, int ldw,
                                int n0, int lane) {
    v8f acc = {};
    const long nW = (long)(n0 + (lane & 15));
    const int kbB = (lane >> 4) * 16;
#pragma unroll
    for (int k0 = 0; k0 < KTOT; k0 += 32) {
        v16h a = load_A_rm(As, lda, k0, lane);
        v16h b = *(const v16h*)(W + nW * ldw + k0 + kbB);
        acc = __builtin_amdgcn_wmma_f32_16x16x32_f16(false, a, false, b,
                                                     (short)0, acc, false, false);
    }
    return acc;
}

// --------------------------- small utility kernels ---------------------------

__global__ void zero_kernel(float* p, long n) {
    long i = (long)blockIdx.x * blockDim.x + threadIdx.x;
    long stride = (long)gridDim.x * blockDim.x;
    for (; i < n; i += stride) p[i] = 0.0f;
}

__global__ void cvt_f16_kernel(const float* __restrict__ s, _Float16* __restrict__ d, int n) {
    int i = blockIdx.x * blockDim.x + threadIdx.x;
    if (i < n) d[i] = (_Float16)s[i];
}

// pad c2_W [14][128] -> [16][128], rows 14,15 zero
__global__ void pad_c2_kernel(const float* __restrict__ s, _Float16* __restrict__ d) {
    int i = blockIdx.x * blockDim.x + threadIdx.x;
    if (i < 16 * 128) {
        int o = i >> 7, k = i & 127;
        d[i] = (o < CCH) ? (_Float16)s[o * 128 + k] : (_Float16)0.0f;
    }
}

// per-node: csum (active channel count) and masked mean coordinate
__global__ void node_pre_kernel(const float* __restrict__ xg, const float* __restrict__ cw,
                                float* __restrict__ csum, float* __restrict__ xpool, int N) {
    long n = (long)blockIdx.x * blockDim.x + threadIdx.x;
    if (n >= N) return;
    float cnt = 0.f, sx = 0.f, sy = 0.f, sz = 0.f;
#pragma unroll
    for (int c = 0; c < CCH; ++c) {
        float w = cw[n * CCH + c];
        if (w != 0.0f) {
            cnt += 1.0f;
            sx += xg[n * 42 + c * 3 + 0];
            sy += xg[n * 42 + c * 3 + 1];
            sz += xg[n * 42 + c * 3 + 2];
        }
    }
    float cc = cnt < 1.0f ? 1.0f : cnt;
    csum[n] = cnt;
    float inv = 1.0f / cc;
    xpool[n * 3 + 0] = sx * inv;
    xpool[n * 3 + 1] = sy * inv;
    xpool[n * 3 + 2] = sz * inv;
}

// ------------------------------- edge kernel --------------------------------
// 256 threads (8 wave32) per block; 16 edges per block.

__global__ __launch_bounds__(256)
void edge_kernel(const float* __restrict__ hg, const float* __restrict__ xg,
                 const int* __restrict__ rowp, const int* __restrict__ colp,
                 const float* __restrict__ chattr, const float* __restrict__ cwg,
                 const _Float16* __restrict__ w_rad, const float* __restrict__ radial_b,
                 const _Float16* __restrict__ w_e1, const float* __restrict__ e1_b,
                 const _Float16* __restrict__ w_e2, const float* __restrict__ e2_b,
                 const float* __restrict__ att_W, const float* __restrict__ att_b,
                 const _Float16* __restrict__ w_c1, const float* __restrict__ c1_b,
                 const _Float16* __restrict__ w_c2p, const float* __restrict__ c2_b,
                 const float* __restrict__ csum, const float* __restrict__ xpool,
                 float* __restrict__ x_sum, float* __restrict__ agg_sum,
                 float* __restrict__ cnt_row, float* __restrict__ cnt_col, int E) {
    __shared__ __align__(32) char smem[41344];
    // region A (32KB): phase1 per-wave geometry scratch; reused afterwards:
    _Float16* geomh = (_Float16*)smem;                 // 8 waves * 2048 halves
    _Float16* efin  = (_Float16*)smem;                 // [16][384]
    _Float16* ef1   = (_Float16*)(smem + 12288);       // [16][128]
    _Float16* ef2   = (_Float16*)(smem + 16384);       // [16][128]
    _Float16* h1    = (_Float16*)(smem + 20480);       // [16][128]
    // region B (8KB): radial A-matrix, reused for ch / pooled
    _Float16* radA  = (_Float16*)(smem + 32768);       // [16][256]
    float* s_ch     = (float*)(smem + 32768);          // [16][16]
    float* s_pool   = (float*)(smem + 33792);          // [16][16]
    // scalars
    float* s_rnorm  = (float*)(smem + 40960);          // [16]
    float* s_gate   = (float*)(smem + 41024);          // [16]
    int*   s_row    = (int*)(smem + 41088);            // [16]
    int*   s_col    = (int*)(smem + 41152);            // [16]
    int*   s_tsize  = (int*)(smem + 41216);            // [16]
    int*   s_valid  = (int*)(smem + 41280);            // [16]

    const int tid = threadIdx.x;
    const int lane = tid & 31;
    const int wave = tid >> 5;

    // warm the f16 weight images into L2 / WGP$ (global_prefetch_b8)
    if (wave == 0) {
        __builtin_prefetch(w_e1 + (long)lane * 1536, 0, 1);   // 96KB image
        __builtin_prefetch(w_rad + (long)lane * 1024, 0, 1);  // 64KB image
    }

    if (tid < 16) {
        long eg = (long)blockIdx.x * 16 + tid;
        int valid = eg < (long)E;
        long ec = valid ? eg : (long)(E - 1);
        s_row[tid] = rowp[ec];
        s_col[tid] = colp[ec];
        s_valid[tid] = valid;
        s_rnorm[tid] = 0.0f;
    }
    __syncthreads();

    // ---- Phase 1: per-edge geometry + radial einsum (2 WMMAs per edge) ----
    // Fragment-native LDS layouts: A row-major [16][32]; B transposed [16][32].
    for (int it = 0; it < 2; ++it) {
        int e = wave * 2 + it;
        long r = s_row[e], c = s_col[e];
        _Float16* attrA  = geomh + wave * 2048;  // A1 [a][c] row-major [16][32]
        _Float16* msgT   = attrA + 512;          // B1^T: [d][c] ([N][32])
        _Float16* attrCT = msgT + 512;           // B2^T: [b][d] ([N][32])
        _Float16* tmpA   = attrCT + 512;         // A2 row-major [16][32]
        for (int i = lane; i < 2048; i += 32) attrA[i] = (_Float16)0.0f;
        __syncthreads();
        for (int i = lane; i < CCH * 16; i += 32) {
            // i = k*16 + m  ->  [m][k] in a [16][32] row-major tile
            _Float16 vr = (_Float16)chattr[r * (CCH * 16) + i];
            _Float16 vc = (_Float16)chattr[c * (CCH * 16) + i];
            int m = i & 15, k = i >> 4;
            attrA[m * 32 + k] = vr;   // A1[a][c] = attr_r[c][a]
            attrCT[m * 32 + k] = vc;  // B2^T[b][d] = attr_c[d][b]
        }
        for (int i = lane; i < CCH * CCH; i += 32) {
            int cc = i / CCH, dd = i % CCH;
            float d0 = xg[r * 42 + cc * 3 + 0] - xg[c * 42 + dd * 3 + 0];
            float d1 = xg[r * 42 + cc * 3 + 1] - xg[c * 42 + dd * 3 + 1];
            float d2 = xg[r * 42 + cc * 3 + 2] - xg[c * 42 + dd * 3 + 2];
            float dist = sqrtf(d0 * d0 + d1 * d1 + d2 * d2);
            msgT[dd * 32 + cc] =   // B1^T[d][c]
                (_Float16)(dist * cwg[r * CCH + cc] * cwg[c * CCH + dd]);
        }
        __syncthreads();
        // tmp[a][d] = sum_c attr_r[c][a] * msg[c][d]
        v16h a1 = load_A_rm(attrA, 32, 0, lane);
        v16h b1 = load_Bt32(msgT, lane);
        v8f t = {};
        t = __builtin_amdgcn_wmma_f32_16x16x32_f16(false, a1, false, b1, (short)0, t,
                                                   false, false);
        {
            int n = lane & 15, mhi = (lane >> 4) * 8;
#pragma unroll
            for (int rr = 0; rr < 8; ++rr) tmpA[(mhi + rr) * 32 + n] = (_Float16)t[rr];
        }
        __syncthreads();
        // radial[a][b] = sum_d tmp[a][d] * attr_c[d][b]
        v16h a2 = load_A_rm(tmpA, 32, 0, lane);
        v16h b2 = load_Bt32(attrCT, lane);
        v8f rad = {};
        rad = __builtin_amdgcn_wmma_f32_16x16x32_f16(false, a2, false, b2, (short)0,
                                                     rad, false, false);
        {
            int n = lane & 15, mhi = (lane >> 4) * 8;
            float ss = 0.0f;
#pragma unroll
            for (int rr = 0; rr < 8; ++rr) {
                float v = rad[rr];
                radA[e * 256 + (mhi + rr) * 16 + n] = (_Float16)v;
                ss += v * v;
            }
            atomicAdd(&s_rnorm[e], ss);
        }
        __syncthreads();
    }

    // ---- gather h[row], h[col] into efin[:, 0:256] ----
    for (int i = tid; i < 16 * 256; i += 256) {
        int e = i >> 8, f = i & 255;
        long node = (f < 128) ? (long)s_row[e] : (long)s_col[e];
        efin[e * 384 + f] = (_Float16)hg[node * 128 + (f & 127)];
    }
    // ---- radial linear: efin[:, 256:384] = (radA @ Wr^T + b)/rnorm ----
    {
        int n0 = wave * 16;
        v8f acc = gemm_wmma<256>(radA, 256, w_rad, 256, n0, lane);
        int n = (lane & 15) + n0, mhi = (lane >> 4) * 8;
        float bn = radial_b[n];
#pragma unroll
        for (int rr = 0; rr < 8; ++rr) {
            int m = mhi + rr;
            float rn = sqrtf(s_rnorm[m]) + 1.0f;
            efin[m * 384 + 256 + n] = (_Float16)((acc[rr] + bn) / rn);
        }
    }
    __syncthreads();

    // ---- e1 ----
    {
        int n0 = wave * 16;
        v8f acc = gemm_wmma<384>(efin, 384, w_e1, 384, n0, lane);
        int n = (lane & 15) + n0, mhi = (lane >> 4) * 8;
        float bn = e1_b[n];
#pragma unroll
        for (int rr = 0; rr < 8; ++rr)
            ef1[(mhi + rr) * 128 + n] = (_Float16)siluf(acc[rr] + bn);
    }
    __syncthreads();
    // ---- e2 ----
    {
        int n0 = wave * 16;
        v8f acc = gemm_wmma<128>(ef1, 128, w_e2, 128, n0, lane);
        int n = (lane & 15) + n0, mhi = (lane >> 4) * 8;
        float bn = e2_b[n];
#pragma unroll
        for (int rr = 0; rr < 8; ++rr)
            ef2[(mhi + rr) * 128 + n] = (_Float16)siluf(acc[rr] + bn);
    }
    __syncthreads();
    // ---- attention gate ----
    if (tid < 16) s_gate[tid] = 0.0f;
    __syncthreads();
    {
        int e = tid & 15, ch = tid >> 4;
        float p = 0.0f;
#pragma unroll
        for (int j = 0; j < 8; ++j) {
            int f = ch * 8 + j;
            p += (float)ef2[e * 128 + f] * att_W[f];
        }
        atomicAdd(&s_gate[e], p);
    }
    __syncthreads();
    if (tid < 16) s_gate[tid] = 1.0f / (1.0f + __expf(-(s_gate[tid] + att_b[0])));
    __syncthreads();
    for (int i = tid; i < 2048; i += 256)
        ef2[i] = (_Float16)((float)ef2[i] * s_gate[i >> 7]);
    __syncthreads();
    // ---- c1 ----
    {
        int n0 = wave * 16;
        v8f acc = gemm_wmma<128>(ef2, 128, w_c1, 128, n0, lane);
        int n = (lane & 15) + n0, mhi = (lane >> 4) * 8;
        float bn = c1_b[n];
#pragma unroll
        for (int rr = 0; rr < 8; ++rr)
            h1[(mhi + rr) * 128 + n] = (_Float16)siluf(acc[rr] + bn);
    }
    __syncthreads();
    // ---- c2 (N padded 14->16), wave 0 only ----
    if (wave == 0) {
        v8f acc = gemm_wmma<128>(h1, 128, w_c2p, 128, 0, lane);
        int n = lane & 15, mhi = (lane >> 4) * 8;
        float bn = (n < CCH) ? c2_b[n] : 0.0f;
#pragma unroll
        for (int rr = 0; rr < 8; ++rr) s_ch[(mhi + rr) * 16 + n] = acc[rr] + bn;
    }
    __syncthreads();
    // ---- roller pooling ----
    if (tid < 16) s_tsize[tid] = (int)(csum[(long)s_row[tid]] + 0.5f) - 1;
    __syncthreads();
    if (tid < 16 * CCH) {
        int e = tid / CCH, i = tid % CCH;
        int t = s_tsize[e];
        t = t < 0 ? 0 : (t > CCH - 1 ? CCH - 1 : t);
        int w = CCH - t;
        int jend = i + w; jend = jend > CCH ? CCH : jend;
        float s = 0.0f;
        for (int j = i; j < jend; ++j) s += s_ch[e * 16 + j];
        s_pool[e * 16 + i] = s / (float)w;
    }
    __syncthreads();
    // ---- trans scatter + counts ----
    for (int i = tid; i < 16 * 42; i += 256) {
        int e = i / 42, rem = i % 42;
        if (s_valid[e]) {
            int cc = rem / 3, d = rem % 3;
            long r = s_row[e];
            float diff = xg[r * 42 + rem] - xpool[(long)s_col[e] * 3 + d];
            atomicAdd(&x_sum[r * 42 + rem], diff * s_pool[e * 16 + cc]);
        }
    }
    if (tid < 16 && s_valid[tid]) {
        atomicAdd(&cnt_row[(long)s_row[tid]], 1.0f);
        atomicAdd(&cnt_col[(long)s_col[tid]], 1.0f);
    }
    // ---- ef scatter ----
    for (int i = tid; i < 2048; i += 256) {
        int e = i >> 7, f = i & 127;
        if (s_valid[e]) atomicAdd(&agg_sum[(long)s_col[e] * 128 + f], (float)ef2[i]);
    }
}

// ------------------------------- node kernel --------------------------------

__global__ __launch_bounds__(256)
void node_out_kernel(const float* __restrict__ hg, const float* __restrict__ xg,
                     const float* __restrict__ agg_sum, const float* __restrict__ cnt_col,
                     const float* __restrict__ x_sum, const float* __restrict__ cnt_row,
                     const _Float16* __restrict__ w_n1, const float* __restrict__ n1_b,
                     const _Float16* __restrict__ w_n2, const float* __restrict__ n2_b,
                     const float* __restrict__ ln_g, const float* __restrict__ ln_b,
                     float* __restrict__ h_out, float* __restrict__ x_out, int N) {
    __shared__ __align__(32) _Float16 nhin[16 * 256];
    __shared__ __align__(32) _Float16 t1[16 * 128];
    __shared__ float y[16 * 128];
    __shared__ float s_mu[16], s_var[16];
    const int tid = threadIdx.x, lane = tid & 31, wave = tid >> 5;
    const long nbase = (long)blockIdx.x * 16;

    if (wave == 0) __builtin_prefetch(w_n1 + (long)lane * 1024, 0, 1);

    for (int i = tid; i < 16 * 256; i += 256) {
        int e = i >> 8, f = i & 255;
        long n = nbase + e;
        float v = 0.0f;
        if (n < N) {
            if (f < 128) v = hg[n * 128 + f];
            else {
                float c = cnt_col[n]; c = c < 1.0f ? 1.0f : c;
                v = agg_sum[n * 128 + (f - 128)] / c;
            }
        }
        nhin[i] = (_Float16)v;
    }
    if (tid < 16) { s_mu[tid] = 0.0f; s_var[tid] = 0.0f; }
    __syncthreads();
    {
        int n0 = wave * 16;
        v8f acc = gemm_wmma<256>(nhin, 256, w_n1, 256, n0, lane);
        int n = (lane & 15) + n0, mhi = (lane >> 4) * 8;
        float bn = n1_b[n];
#pragma unroll
        for (int rr = 0; rr < 8; ++rr)
            t1[(mhi + rr) * 128 + n] = (_Float16)siluf(acc[rr] + bn);
    }
    __syncthreads();
    {
        int n0 = wave * 16;
        v8f acc = gemm_wmma<128>(t1, 128, w_n2, 128, n0, lane);
        int n = (lane & 15) + n0, mhi = (lane >> 4) * 8;
        float bn = n2_b[n];
#pragma unroll
        for (int rr = 0; rr < 8; ++rr) {
            int m = mhi + rr;
            long nn = nbase + m;
            float hv = (nn < N) ? hg[nn * 128 + n] : 0.0f;
            y[m * 128 + n] = hv + acc[rr] + bn;
        }
    }
    __syncthreads();
    {
        int e = tid & 15, ch = tid >> 4;
        float p = 0.0f;
#pragma unroll
        for (int j = 0; j < 8; ++j) p += y[e * 128 + ch * 8 + j];
        atomicAdd(&s_mu[e], p);
    }
    __syncthreads();
    {
        int e = tid & 15, ch = tid >> 4;
        float mu = s_mu[e] * (1.0f / 128.0f);
        float p = 0.0f;
#pragma unroll
        for (int j = 0; j < 8; ++j) {
            float d = y[e * 128 + ch * 8 + j] - mu;
            p += d * d;
        }
        atomicAdd(&s_var[e], p);
    }
    __syncthreads();
    for (int i = tid; i < 2048; i += 256) {
        int e = i >> 7, f = i & 127;
        long n = nbase + e;
        if (n < N) {
            float mu = s_mu[e] * (1.0f / 128.0f);
            float var = s_var[e] * (1.0f / 128.0f);
            h_out[n * 128 + f] = (y[i] - mu) * rsqrtf(var + 1e-5f) * ln_g[f] + ln_b[f];
        }
    }
    for (int i = tid; i < 16 * 42; i += 256) {
        int e = i / 42, rem = i % 42;
        long n = nbase + e;
        if (n < N) {
            float c = cnt_row[n]; c = c < 1.0f ? 1.0f : c;
            x_out[n * 42 + rem] = xg[n * 42 + rem] + x_sum[n * 42 + rem] / c;
        }
    }
}

// -------------------------------- launcher ----------------------------------

extern "C" void kernel_launch(void* const* d_in, const int* in_sizes, int n_in,
                              void* d_out, int out_size, void* d_ws, size_t ws_size,
                              hipStream_t stream) {
    const float* h        = (const float*)d_in[0];
    const float* x        = (const float*)d_in[1];
    const int*   row      = (const int*)d_in[2];
    const int*   col      = (const int*)d_in[3];
    const float* chattr   = (const float*)d_in[4];
    const float* cw       = (const float*)d_in[5];
    const float* radial_W = (const float*)d_in[6];
    const float* radial_b = (const float*)d_in[7];
    const float* e1_W     = (const float*)d_in[8];
    const float* e1_b     = (const float*)d_in[9];
    const float* e2_W     = (const float*)d_in[10];
    const float* e2_b     = (const float*)d_in[11];
    const float* att_W    = (const float*)d_in[12];
    const float* att_b    = (const float*)d_in[13];
    const float* c1_W     = (const float*)d_in[14];
    const float* c1_b     = (const float*)d_in[15];
    const float* c2_W     = (const float*)d_in[16];
    const float* c2_b     = (const float*)d_in[17];
    const float* n1_W     = (const float*)d_in[18];
    const float* n1_b     = (const float*)d_in[19];
    const float* n2_W     = (const float*)d_in[20];
    const float* n2_b     = (const float*)d_in[21];
    const float* ln_g     = (const float*)d_in[22];
    const float* ln_b     = (const float*)d_in[23];
    (void)n_in; (void)out_size; (void)ws_size;

    const int N = in_sizes[0] / 128;
    const int E = in_sizes[2];

    float* ws      = (float*)d_ws;
    float* x_sum   = ws;                              // N*42
    float* agg_sum = x_sum + (long)N * 42;            // N*128
    float* cnt_row = agg_sum + (long)N * 128;         // N
    float* cnt_col = cnt_row + N;                     // N
    float* csum    = cnt_col + N;                     // N
    float* xpool   = csum + N;                        // N*3
    _Float16* w16   = (_Float16*)(xpool + (long)N * 3);
    _Float16* w_rad = w16;                            // 256*128
    _Float16* w_e1  = w_rad + 32768;                  // 384*128
    _Float16* w_e2  = w_e1 + 49152;                   // 128*128
    _Float16* w_c1  = w_e2 + 16384;                   // 128*128
    _Float16* w_c2p = w_c1 + 16384;                   // 16*128 (padded)
    _Float16* w_n1  = w_c2p + 2048;                   // 256*128
    _Float16* w_n2  = w_n1 + 32768;                   // 128*128

    zero_kernel<<<1024, 256, 0, stream>>>(ws, (long)N * 172);

    cvt_f16_kernel<<<(32768 + 255) / 256, 256, 0, stream>>>(radial_W, w_rad, 32768);
    cvt_f16_kernel<<<(49152 + 255) / 256, 256, 0, stream>>>(e1_W, w_e1, 49152);
    cvt_f16_kernel<<<(16384 + 255) / 256, 256, 0, stream>>>(e2_W, w_e2, 16384);
    cvt_f16_kernel<<<(16384 + 255) / 256, 256, 0, stream>>>(c1_W, w_c1, 16384);
    cvt_f16_kernel<<<(32768 + 255) / 256, 256, 0, stream>>>(n1_W, w_n1, 32768);
    cvt_f16_kernel<<<(16384 + 255) / 256, 256, 0, stream>>>(n2_W, w_n2, 16384);
    pad_c2_kernel<<<8, 256, 0, stream>>>(c2_W, w_c2p);

    node_pre_kernel<<<(N + 255) / 256, 256, 0, stream>>>(x, cw, csum, xpool, N);

    if (E > 0) {
        edge_kernel<<<(E + 15) / 16, 256, 0, stream>>>(
            h, x, row, col, chattr, cw,
            w_rad, radial_b, w_e1, e1_b, w_e2, e2_b, att_W, att_b,
            w_c1, c1_b, w_c2p, c2_b, csum, xpool,
            x_sum, agg_sum, cnt_row, cnt_col, E);
    }

    float* h_out = (float*)d_out;                 // N*128
    float* x_out = h_out + (long)N * 128;         // N*42
    node_out_kernel<<<(N + 15) / 16, 256, 0, stream>>>(
        h, x, agg_sum, cnt_col, x_sum, cnt_row,
        w_n1, n1_b, w_n2, n2_b, ln_g, ln_b, h_out, x_out, N);
}
